// SimpleBlock2d_86388972192315
// MI455X (gfx1250) — matrix operand order repaired and checked by reading, not compiled, not run on hardware
//
#include <hip/hip_runtime.h>
#include <math.h>
#include <stdint.h>

typedef __attribute__((ext_vector_type(16))) _Float16 v16h;
typedef __attribute__((ext_vector_type(8)))  float    v8f;

#define Bb   16
#define Cc   32
#define Hh   128
#define Ww   128
#define HW   16384
#define M1m  12
#define M2m  12
#define Ee   128
#define HIDh 64
#define TWOPI 6.2831853071795864f

__device__ __forceinline__ float gelu_exact(float v) {
    return 0.5f * v * (1.0f + erff(v * 0.70710678118654752f));
}

// ---- CDNA5 async global->LDS copy (ASYNCcnt-tracked, ISA ch.10 async ops) ----
__device__ __forceinline__ void async_load_dword(uint32_t lds_off, const float* gaddr) {
    asm volatile("global_load_async_to_lds_b32 %0, %1, off"
                 :: "v"(lds_off), "v"((uint64_t)(uintptr_t)gaddr)
                 : "memory");
}
__device__ __forceinline__ void wait_async0() {
    asm volatile("s_wait_asynccnt 0x0" ::: "memory");
}

// -------------------- lifting: h = x @ fc0_w^T + b --------------------
__global__ void lift_kernel(const float* __restrict__ x, const float* __restrict__ w,
                            const float* __restrict__ bia, float* __restrict__ h) {
    int idx = blockIdx.x * blockDim.x + threadIdx.x;          // B*C*HW = 8388608
    int b   = idx >> 19;
    int rem = idx & ((1 << 19) - 1);
    int o   = rem >> 14;
    int p   = rem & (HW - 1);
    const float* xb = x + ((size_t)b * 3) * HW + p;
    float acc = bia[o];
    acc += xb[0]      * w[o * 3 + 0];
    acc += xb[HW]     * w[o * 3 + 1];
    acc += xb[2 * HW] * w[o * 3 + 2];
    h[idx] = acc;
}

// ----- DFT along y as WMMA: [rows=BCH,128] x [128, 12cos|12sin] -----------
// Twiddle tables padded to 16 columns (zeros in 12..15) -> branchless B frags.
__global__ void dft_y_kernel(const float* __restrict__ h, float* __restrict__ G) {
    __shared__ float tc16[Ww * 16], ts16[Ww * 16];            // [y][ky(16)]
    for (int k = threadIdx.x; k < Ww * 16; k += blockDim.x) {
        int y = k >> 4, ky = k & 15;
        float ang = TWOPI * (float)(ky * y) / (float)Ww;
        float c = (ky < M2m) ? cosf(ang) : 0.f;
        float s = (ky < M2m) ? sinf(ang) : 0.f;
        tc16[k] = c; ts16[k] = s;
    }
    __syncthreads();
    int wv   = blockIdx.x * 8 + (threadIdx.x >> 5);           // 4096 waves
    int lane = threadIdx.x & 31;
    int half = lane >> 4;
    int mrow = lane & 15;
    int r0   = wv * 16;                                       // row base in [0,65536)
    const float* rowp = h + (size_t)(r0 + mrow) * Ww;
    v8f accR = {}, accI = {};
    for (int kk = 0; kk < 4; ++kk) {
        v16h A;
#pragma unroll
        for (int e = 0; e < 16; ++e) {
            int K = half * 16 + e;
            A[e] = (_Float16)rowp[kk * 32 + K];
        }
        v16h Bc, Bs;                                          // col = mrow = ky
#pragma unroll
        for (int e = 0; e < 16; ++e) {
            int y = kk * 32 + e + half * 16;
            Bc[e] = (_Float16)tc16[(y << 4) + mrow];
            Bs[e] = (_Float16)(-ts16[(y << 4) + mrow]);
        }
        accR = __builtin_amdgcn_wmma_f32_16x16x32_f16(false, A, false, Bc, (short)0, accR, false, false);
        accI = __builtin_amdgcn_wmma_f32_16x16x32_f16(false, A, false, Bs, (short)0, accI, false, false);
    }
    if (mrow < M2m) {                                         // D col N = mrow = ky
        const float s = 1.0f / (float)(Hh * Ww);              // norm="forward"
#pragma unroll
        for (int r = 0; r < 8; ++r) {
            int row = r0 + r + half * 8;
            G[(size_t)row * (2 * M2m) + 2 * mrow]     = accR[r] * s;
            G[(size_t)row * (2 * M2m) + 2 * mrow + 1] = accI[r] * s;
        }
    }
}

// -------------------- DFT along x (complex, keep 12 modes) ----------------
__global__ void dft_x_kernel(const float* __restrict__ G,
                             float* __restrict__ xftr, float* __restrict__ xfti) {
    __shared__ float tc[Hh * M1m], ts[Hh * M1m];
    for (int k = threadIdx.x; k < Hh * M1m; k += blockDim.x) {
        int xx = k / M1m, kx = k % M1m;
        float ang = TWOPI * (float)(kx * xx) / (float)Hh;
        tc[k] = cosf(ang); ts[k] = sinf(ang);
    }
    __syncthreads();
    int idx = blockIdx.x * blockDim.x + threadIdx.x;          // 73728
    int bc  = idx / (M1m * M2m);
    int rem = idx % (M1m * M2m);
    int kx  = rem / M2m;
    int ky  = rem % M2m;
    const float* g = G + ((size_t)bc * Hh) * (2 * M2m) + 2 * ky;
    float accr = 0.f, acci = 0.f;
    for (int xx = 0; xx < Hh; ++xx) {
        float Gr = g[xx * 2 * M2m], Gi = g[xx * 2 * M2m + 1];
        float c = tc[xx * M1m + kx], s = ts[xx * M1m + kx];   // e^{-i ang} = (c,-s)
        accr += Gr * c + Gi * s;
        acci += Gi * c - Gr * s;
    }
    xftr[idx] = accr; xfti[idx] = acci;
}

// -------------------- phi_p / phi_n from emb ------------------------------
__global__ void phi_kernel(const float* __restrict__ emb,
                           const float* __restrict__ Arp, const float* __restrict__ Aip,
                           const float* __restrict__ Arn, const float* __restrict__ Ain,
                           float* ppr, float* ppi, float* pnr, float* pni) {
    int idx = blockIdx.x * blockDim.x + threadIdx.x;          // 2304
    int b = idx / (M1m * M2m);
    int m = idx % (M1m * M2m);
    const float* e  = emb + b * Ee;
    const float* a1 = Arp + m * Ee;
    const float* a2 = Aip + m * Ee;
    const float* a3 = Arn + m * Ee;
    const float* a4 = Ain + m * Ee;
    float s1 = 0.f, s2 = 0.f, s3 = 0.f, s4 = 0.f;
    for (int k = 0; k < Ee; ++k) {
        float ev = e[k];
        s1 += a1[k] * ev; s2 += a2[k] * ev; s3 += a3[k] * ev; s4 += a4[k] * ev;
    }
    ppr[idx] = s1; ppi[idx] = s2; pnr[idx] = s3; pni[idx] = s4;
}

// -------------------- time MLP -> gamma(=1+g), beta -----------------------
__global__ void timemlp_kernel(const float* __restrict__ emb,
                               const float* __restrict__ w1, const float* __restrict__ b1,
                               const float* __restrict__ w2, const float* __restrict__ b2,
                               float* __restrict__ gam, float* __restrict__ bet) {
    __shared__ float hid[HIDh];
    int b = blockIdx.x;
    int j = threadIdx.x;                                      // 64 threads
    float acc = b1[j];
    const float* e = emb + b * Ee;
    const float* w1r = w1 + j * Ee;
    for (int k = 0; k < Ee; ++k) acc += e[k] * w1r[k];
    hid[j] = acc / (1.0f + expf(-acc));                       // silu
    __syncthreads();
    float acc2 = b2[j];
    const float* w2r = w2 + j * HIDh;
    for (int k = 0; k < HIDh; ++k) acc2 += hid[k] * w2r[k];
    if (j < Cc) gam[b * Cc + j] = 1.0f + acc2;
    else        bet[b * Cc + (j - Cc)] = acc2;
}

// -------- per-mode complex [B,Ci]x[Ci,Co] matmul via WMMA + phi modulation --
__global__ void mode_matmul_kernel(const float* __restrict__ xftr, const float* __restrict__ xfti,
                                   const float* __restrict__ w,
                                   const float* __restrict__ ppr, const float* __restrict__ ppi,
                                   const float* __restrict__ pnr, const float* __restrict__ pni,
                                   float* Spr, float* Spi, float* Snr, float* Sni) {
    int wid  = blockIdx.x * 8 + (threadIdx.x >> 5);           // 288 waves
    int lane = threadIdx.x & 31;
    int mode = wid >> 1;
    int nt   = wid & 1;
    int half = lane >> 4;
    int mrow = lane & 15;
    int kbA  = half * 8;
    v16h Ar, Ai, An;
#pragma unroll
    for (int e = 0; e < 16; ++e) {
        int k = kbA + e + ((e >= 8) ? 8 : 0);
        size_t ix = ((size_t)(mrow * Cc + k)) * (M1m * M2m) + mode;
        float vr = xftr[ix], vi = xfti[ix];
        Ar[e] = (_Float16)vr; Ai[e] = (_Float16)vi; An[e] = (_Float16)(-vi);
    }
    int o = nt * 16 + mrow;
    v16h Br, Bi;
#pragma unroll
    for (int e = 0; e < 16; ++e) {
        int ci = half * 8 + e + ((e >= 8) ? 8 : 0);
        size_t ix = (((size_t)(ci * Cc + o)) * (M1m * M2m) + mode) * 2;
        Br[e] = (_Float16)w[ix]; Bi[e] = (_Float16)w[ix + 1];
    }
    v8f z = {};
    v8f Dr = __builtin_amdgcn_wmma_f32_16x16x32_f16(false, An, false, Bi, (short)0, z, false, false);
    Dr     = __builtin_amdgcn_wmma_f32_16x16x32_f16(false, Ar, false, Br, (short)0, Dr, false, false);
    v8f Di = __builtin_amdgcn_wmma_f32_16x16x32_f16(false, Ai, false, Br, (short)0, z, false, false);
    Di     = __builtin_amdgcn_wmma_f32_16x16x32_f16(false, Ar, false, Bi, (short)0, Di, false, false);
#pragma unroll
    for (int r = 0; r < 8; ++r) {
        int bb = r + half * 8;
        float dr = Dr[r], di = Di[r];
        int pix = bb * (M1m * M2m) + mode;
        float pr = ppr[pix], pi = ppi[pix];
        float nr = pnr[pix], ni = pni[pix];
        size_t ox = ((size_t)(bb * Cc + o)) * (M1m * M2m) + mode;
        Spr[ox] = dr * pr - di * pi;
        Spi[ox] = dr * pi + di * pr;
        Snr[ox] = dr * nr - di * ni;
        Sni[ox] = dr * ni + di * nr;
    }
}

// ---- FiLM 1x1 conv via WMMA; h-tile AND weights async-staged to LDS ------
__global__ void film_conv_kernel(const float* __restrict__ h, const float* __restrict__ cw,
                                 const float* __restrict__ cb, const float* __restrict__ gam,
                                 const float* __restrict__ bet, float* __restrict__ h2) {
    __shared__ float swc[Cc * Cc];                            // 4KB weights
    __shared__ float tile[Cc * 128];                          // 16KB: [ch][128 px]
    int gp0 = blockIdx.x * 128;                               // block pixel base
    int b   = gp0 >> 14;
    int p0  = gp0 & (HW - 1);
    for (int k = threadIdx.x; k < Cc * Cc; k += blockDim.x)
        async_load_dword((uint32_t)(uintptr_t)&swc[k], cw + k);
    for (int k = threadIdx.x; k < Cc * 128; k += blockDim.x) {
        int ch = k >> 7, px = k & 127;                        // consecutive threads -> px
        async_load_dword((uint32_t)(uintptr_t)&tile[k],
                         h + ((size_t)(b * Cc + ch) << 14) + p0 + px);
    }
    wait_async0();
    __syncthreads();

    int wloc = threadIdx.x >> 5;                              // wave in block: 0..7
    int lane = threadIdx.x & 31;
    int half = lane >> 4;
    int mrow = lane & 15;
    int pt   = wloc * 16;                                     // wave pixel base in tile
    int kbA  = half * 8;
    v16h A;
#pragma unroll
    for (int e = 0; e < 16; ++e) {
        int k = kbA + e + ((e >= 8) ? 8 : 0);
        A[e] = (_Float16)tile[(k << 7) + pt + mrow];
    }
    v8f z = {};
#pragma unroll
    for (int nt = 0; nt < 2; ++nt) {
        int o = nt * 16 + mrow;
        v16h Bt;
#pragma unroll
        for (int e = 0; e < 16; ++e) Bt[e] = (_Float16)swc[o * Cc + half * 8 + e + ((e >= 8) ? 8 : 0)];
        v8f D = __builtin_amdgcn_wmma_f32_16x16x32_f16(false, A, false, Bt, (short)0, z, false, false);
        float g = gam[b * Cc + o], bt = bet[b * Cc + o], cbias = cb[o];
#pragma unroll
        for (int r = 0; r < 8; ++r) {
            int pix = p0 + pt + r + half * 8;
            h2[((size_t)(b * Cc + o) << 14) + pix] = (D[r] + cbias) * g + bt;
        }
    }
}

// -------------------- inverse DFT along x (both bands) --------------------
__global__ void inv_x_kernel(const float* __restrict__ Spr, const float* __restrict__ Spi,
                             const float* __restrict__ Snr, const float* __restrict__ Sni,
                             float* __restrict__ T) {
    __shared__ float tc[Hh * 13], ts[Hh * 13];
    for (int k = threadIdx.x; k < Hh * 13; k += blockDim.x) {
        int xx = k / 13, d = k % 13;
        float ang = TWOPI * (float)(d * xx) / (float)Hh;
        tc[k] = cosf(ang); ts[k] = sinf(ang);
    }
    __syncthreads();
    int idx = blockIdx.x * blockDim.x + threadIdx.x;          // 786432
    int bo  = idx / (Hh * M2m);
    int rem = idx % (Hh * M2m);
    int xx  = rem / M2m;
    int ky  = rem % M2m;
    const float* pr = Spr + (size_t)bo * (M1m * M2m) + ky;
    const float* pi = Spi + (size_t)bo * (M1m * M2m) + ky;
    const float* nr = Snr + (size_t)bo * (M1m * M2m) + ky;
    const float* ni = Sni + (size_t)bo * (M1m * M2m) + ky;
    float Tr = 0.f, Ti = 0.f;
#pragma unroll
    for (int j = 0; j < M1m; ++j) {
        float c = tc[xx * 13 + j], s = ts[xx * 13 + j];       // e^{+2pi i j x/H}
        float ar = pr[j * M2m], ai = pi[j * M2m];
        Tr += ar * c - ai * s; Ti += ar * s + ai * c;
        float cn = tc[xx * 13 + (12 - j)], sn = ts[xx * 13 + (12 - j)]; // freq j-12
        float br = nr[j * M2m], bi = ni[j * M2m];
        Tr += br * cn + bi * sn; Ti += bi * cn - br * sn;
    }
    T[(size_t)idx * 2]     = Tr;
    T[(size_t)idx * 2 + 1] = Ti;
}

// ---- inverse rfft along y as WMMA, fused add FiLM branch + optional GELU --
// out[row,y] = Tr . [1,2cos..] + Ti . [0,-2sin..]  (A: K0-11=Tr, K16-27=Ti)
__global__ void inv_y_kernel(const float* __restrict__ T, float* __restrict__ h2,
                             int apply_gelu) {
    __shared__ float tc16[Ww * 16], ts16[Ww * 16];            // [y][ky(16)], padded
    for (int k = threadIdx.x; k < Ww * 16; k += blockDim.x) {
        int y = k >> 4, ky = k & 15;
        float ang = TWOPI * (float)(ky * y) / (float)Ww;
        tc16[k] = (ky < M2m) ? cosf(ang) : 0.f;
        ts16[k] = (ky < M2m) ? sinf(ang) : 0.f;
    }
    __syncthreads();
    int wv   = blockIdx.x * 8 + (threadIdx.x >> 5);           // 4096 waves
    int lane = threadIdx.x & 31;
    int half = lane >> 4;
    int mrow = lane & 15;
    int r0   = wv * 16;                                       // rows = bo*128 + x
    const float* tp = T + (size_t)(r0 + mrow) * (2 * M2m);
    v16h A;
#pragma unroll
    for (int e = 0; e < 16; ++e) {
        int K = half * 16 + e;
        float v = 0.f;
        if (K < M2m)                 v = tp[2 * K];           // Tr[ky]
        else if (K >= 16 && K < 28)  v = tp[2 * (K - 16) + 1];// Ti[ky]
        A[e] = (_Float16)v;
    }
    v8f z = {};
    for (int nt = 0; nt < 8; ++nt) {
        int y = nt * 16 + mrow;                               // B col = D col
        v16h Bt;
#pragma unroll
        for (int e = 0; e < 16; ++e) {
            float v;
            if (half == 0)  v = (e == 0) ? 1.0f :  2.0f * tc16[(y << 4) + e];
            else            v = (e == 0) ? 0.0f : -2.0f * ts16[(y << 4) + e];
            Bt[e] = (_Float16)v;
        }
        v8f D = __builtin_amdgcn_wmma_f32_16x16x32_f16(false, A, false, Bt, (short)0, z, false, false);
#pragma unroll
        for (int r = 0; r < 8; ++r) {
            int row = r0 + r + half * 8;
            size_t idx = (size_t)row * Ww + y;
            float v = D[r] + h2[idx];
            if (apply_gelu) v = gelu_exact(v);
            h2[idx] = v;
        }
    }
}

// ---- head: fc1 (WMMA, h-tile + weights async-staged) + gelu + fc2 --------
__global__ void head_kernel(const float* __restrict__ h,
                            const float* __restrict__ w1, const float* __restrict__ b1,
                            const float* __restrict__ w2, const float* __restrict__ b2,
                            float* __restrict__ out) {
    __shared__ float hid[64 * 128];                           // 32KB
    __shared__ float sw1[128 * Cc];                           // 16KB
    __shared__ float tile[Cc * 64];                           // 8KB: [ch][64 px]
    __shared__ float sw2[128];
    int gp0 = blockIdx.x * 64;
    int b   = gp0 >> 14;
    int p0  = gp0 & (HW - 1);
    for (int k = threadIdx.x; k < 128 * Cc; k += blockDim.x)
        async_load_dword((uint32_t)(uintptr_t)&sw1[k], w1 + k);
    for (int k = threadIdx.x; k < Cc * 64; k += blockDim.x) {
        int ch = k >> 6, px = k & 63;
        async_load_dword((uint32_t)(uintptr_t)&tile[k],
                         h + ((size_t)(b * Cc + ch) << 14) + p0 + px);
    }
    for (int k = threadIdx.x; k < 128; k += blockDim.x)
        async_load_dword((uint32_t)(uintptr_t)&sw2[k], w2 + k);
    wait_async0();
    __syncthreads();

    int warp = threadIdx.x >> 5;                              // block=128 -> 4 waves
    int lane = threadIdx.x & 31;
    int half = lane >> 4;
    int mrow = lane & 15;
    int kbA  = half * 8;
    v16h A;
#pragma unroll
    for (int e = 0; e < 16; ++e) {
        int k = kbA + e + ((e >= 8) ? 8 : 0);
        A[e] = (_Float16)tile[(k << 6) + warp * 16 + mrow];
    }
    v8f z = {};
    for (int nt = 0; nt < 8; ++nt) {
        int o = nt * 16 + mrow;
        v16h Bt;
#pragma unroll
        for (int e = 0; e < 16; ++e) Bt[e] = (_Float16)sw1[o * Cc + half * 8 + e + ((e >= 8) ? 8 : 0)];
        v8f D = __builtin_amdgcn_wmma_f32_16x16x32_f16(false, A, false, Bt, (short)0, z, false, false);
        float bias = b1[o];
#pragma unroll
        for (int r = 0; r < 8; ++r) {
            float v = gelu_exact(D[r] + bias);
            int ploc = warp * 16 + r + half * 8;
            hid[ploc * 128 + o] = v;
        }
    }
    __syncthreads();
    if (threadIdx.x < 64) {
        int ploc = threadIdx.x;
        const float* hv = &hid[ploc * 128];
        float acc = b2[0];
        for (int j = 0; j < 128; ++j) acc += hv[j] * sw2[j];
        out[(size_t)blockIdx.x * 64 + ploc] = acc;            // [B,1,H,W] flat
    }
}

// ==========================================================================
extern "C" void kernel_launch(void* const* d_in, const int* in_sizes, int n_in,
                              void* d_out, int out_size, void* d_ws, size_t ws_size,
                              hipStream_t stream) {
    const float* x      = (const float*)d_in[0];
    const float* emb    = (const float*)d_in[1];
    const float* fc0_w  = (const float*)d_in[2];
    const float* fc0_b  = (const float*)d_in[3];
    const float* sc_w   = (const float*)d_in[4];
    const float* sc_Arp = (const float*)d_in[5];
    const float* sc_Aip = (const float*)d_in[6];
    const float* sc_Arn = (const float*)d_in[7];
    const float* sc_Ain = (const float*)d_in[8];
    const float* tm_w1  = (const float*)d_in[9];
    const float* tm_b1  = (const float*)d_in[10];
    const float* tm_w2  = (const float*)d_in[11];
    const float* tm_b2  = (const float*)d_in[12];
    const float* cv_w   = (const float*)d_in[13];
    const float* cv_b   = (const float*)d_in[14];
    const float* fc1_w  = (const float*)d_in[15];
    const float* fc1_b  = (const float*)d_in[16];
    const float* fc2_w  = (const float*)d_in[17];
    const float* fc2_b  = (const float*)d_in[18];

    float* ws = (float*)d_ws;
    const size_t HSZ = (size_t)Bb * Cc * HW;                  // 8388608
    float* hA   = ws;                 size_t off = HSZ;
    float* hB   = ws + off;           off += HSZ;
    float* G    = ws + off;           off += (size_t)Bb * Cc * Hh * M2m * 2;
    float* T    = ws + off;           off += (size_t)Bb * Cc * Hh * M2m * 2;
    float* xftr = ws + off;           off += (size_t)Bb * Cc * M1m * M2m;
    float* xfti = ws + off;           off += (size_t)Bb * Cc * M1m * M2m;
    float* Spr  = ws + off;           off += (size_t)Bb * Cc * M1m * M2m;
    float* Spi  = ws + off;           off += (size_t)Bb * Cc * M1m * M2m;
    float* Snr  = ws + off;           off += (size_t)Bb * Cc * M1m * M2m;
    float* Sni  = ws + off;           off += (size_t)Bb * Cc * M1m * M2m;
    float* ppr  = ws + off;           off += (size_t)Bb * M1m * M2m;
    float* ppi  = ws + off;           off += (size_t)Bb * M1m * M2m;
    float* pnr  = ws + off;           off += (size_t)Bb * M1m * M2m;
    float* pni  = ws + off;           off += (size_t)Bb * M1m * M2m;
    float* gam  = ws + off;           off += (size_t)Bb * Cc;
    float* bet  = ws + off;           off += (size_t)Bb * Cc;

    lift_kernel<<<32768, 256, 0, stream>>>(x, fc0_w, fc0_b, hA);

    for (int l = 0; l < 4; ++l) {
        float* hin  = (l & 1) ? hB : hA;
        float* hout = (l & 1) ? hA : hB;
        timemlp_kernel<<<Bb, HIDh, 0, stream>>>(emb,
                                                tm_w1 + (size_t)l * HIDh * Ee,
                                                tm_b1 + (size_t)l * HIDh,
                                                tm_w2 + (size_t)l * 2 * Cc * HIDh,
                                                tm_b2 + (size_t)l * 2 * Cc,
                                                gam, bet);
        dft_y_kernel<<<512, 256, 0, stream>>>(hin, G);        // 4096 WMMA waves
        dft_x_kernel<<<288, 256, 0, stream>>>(G, xftr, xfti);
        phi_kernel<<<9, 256, 0, stream>>>(emb,
                                          sc_Arp + (size_t)l * M1m * M2m * Ee,
                                          sc_Aip + (size_t)l * M1m * M2m * Ee,
                                          sc_Arn + (size_t)l * M1m * M2m * Ee,
                                          sc_Ain + (size_t)l * M1m * M2m * Ee,
                                          ppr, ppi, pnr, pni);
        mode_matmul_kernel<<<36, 256, 0, stream>>>(xftr, xfti,
                                                   sc_w + (size_t)l * Cc * Cc * M1m * M2m * 2,
                                                   ppr, ppi, pnr, pni,
                                                   Spr, Spi, Snr, Sni);
        film_conv_kernel<<<2048, 256, 0, stream>>>(hin,
                                                   cv_w + (size_t)l * Cc * Cc,
                                                   cv_b + (size_t)l * Cc,
                                                   gam, bet, hout);
        inv_x_kernel<<<3072, 256, 0, stream>>>(Spr, Spi, Snr, Sni, T);
        inv_y_kernel<<<512, 256, 0, stream>>>(T, hout, (l < 3) ? 1 : 0);
    }
    // after 4 layers (0:A->B, 1:B->A, 2:A->B, 3:B->A) result is in hA
    head_kernel<<<4096, 128, 0, stream>>>(hA, fc1_w, fc1_b, fc2_w, fc2_b, (float*)d_out);
}
